// SimpleTransformerBlock_84799834292235
// MI455X (gfx1250) — matrix operand order repaired and checked by reading, not compiled, not run on hardware
//
#include <hip/hip_runtime.h>
#include <hip/hip_bf16.h>
#include <cmath>

// ---------------------------------------------------------------------------
// Types for CDNA5 WMMA (wave32): D(16x16 f32) = A(16x32 bf16) x B(32x16 bf16) + C
// ---------------------------------------------------------------------------
typedef __attribute__((ext_vector_type(16))) __bf16 v16bf;
typedef __attribute__((ext_vector_type(8)))  float  v8f;
typedef __attribute__((ext_vector_type(4)))  float  f4;     // 16-byte chunk
typedef __attribute__((ext_vector_type(4)))  unsigned u32x4;
typedef __attribute__((ext_vector_type(8)))  int      i32x8;
typedef __attribute__((ext_vector_type(4)))  int      i32x4;

// Load a 32-byte WMMA fragment (16 bf16) from LDS as two 16B pieces.
__device__ __forceinline__ v16bf ld_frag(const __bf16* p0, const __bf16* p1) {
    struct S { f4 a, b; } s;
    s.a = *(const f4*)p0;
    s.b = *(const f4*)p1;
    return __builtin_bit_cast(v16bf, s);
}

__device__ __forceinline__ v8f wmma_bf16(v16bf a, v16bf b, v8f c) {
    return __builtin_amdgcn_wmma_f32_16x16x32_bf16(false, a, false, b,
                                                   (short)0, c, false, false);
}

// ---------------------------------------------------------------------------
// Tensor Data Mover: async DMA of a 2D bf16 tile (32 elems x 128 rows) from a
// row-major (rows x K) global tensor into LDS.  D# built per CDNA5 ISA §8.3/8.4.
// Issued once per wave (EXEC ignored); completion via TENSORcnt.
// Toolchain note: clang-23 / therock signature is 6 args
//   (u32x4 g0, i32x8 g1, i32x4 g2, i32x4 g3, i32x8 pad, i32 cpol)
// ---------------------------------------------------------------------------
__device__ __forceinline__ void tdm_load_a_tile(const __bf16* gtile,
                                                unsigned lds_off,
                                                int rows, int tensor_k) {
    const unsigned long long ga = (unsigned long long)(size_t)gtile;
    u32x4 g0;
    g0[0] = 1u;                                          // count=1 (valid user D#)
    g0[1] = lds_off;                                     // LDS byte address
    g0[2] = (unsigned)(ga & 0xFFFFFFFFu);                // global_addr[31:0]
    g0[3] = (unsigned)((ga >> 32) & 0x1FFFFFFu)          // global_addr[56:32]
          | (2u << 30);                                  // type=2 ("image")
    const unsigned dim0 = (unsigned)tensor_k;            // tensor_dim0 (elems)
    const unsigned dim1 = 0x7FFFFFFFu;                   // tensor_dim1 (no OOB)
    i32x8 g1;
    g1[0] = (int)(1u << 16);                             // data_size=1 (2 bytes)
    g1[1] = (int)((dim0 & 0xFFFFu) << 16);               // dim0[15:0]
    g1[2] = (int)(((dim0 >> 16) & 0xFFFFu)               // dim0[31:16]
          | ((dim1 & 0xFFFFu) << 16));                   // dim1[15:0]
    g1[3] = (int)(((dim1 >> 16) & 0xFFFFu)               // dim1[31:16]
          | (32u << 16));                                // tile_dim0 = 32
    g1[4] = (int)(unsigned)rows;                         // tile_dim1 ; tile_dim2=0
    g1[5] = (int)dim0;                                   // dim0_stride[31:0] = K
    g1[6] = 0;                                           // stride0 hi / stride1 lo
    g1[7] = 0;
    const i32x4 z4 = {0, 0, 0, 0};
    const i32x8 z8 = {0, 0, 0, 0, 0, 0, 0, 0};
    __builtin_amdgcn_tensor_load_to_lds(g0, g1, z4, z4, z8, 0);
}

// ---------------------------------------------------------------------------
// f32 -> bf16 conversion (weights / activations)
// ---------------------------------------------------------------------------
__global__ void cvt_f32_bf16_kernel(const float* __restrict__ s,
                                    __bf16* __restrict__ d, int n) {
    for (int i = blockIdx.x * blockDim.x + threadIdx.x; i < n;
         i += gridDim.x * blockDim.x)
        d[i] = (__bf16)s[i];
}

// ---------------------------------------------------------------------------
// LayerNorm (one block per row of C=1024), emits bf16 for WMMA consumption
// ---------------------------------------------------------------------------
__global__ __launch_bounds__(256)
void layernorm_bf16_kernel(const float* __restrict__ x,
                           const float* __restrict__ g,
                           const float* __restrict__ b,
                           __bf16* __restrict__ out, int C) {
    const int row = blockIdx.x;
    const float* xr = x + (size_t)row * C;
    float s = 0.f, ss = 0.f;
    for (int i = threadIdx.x; i < C; i += 256) {
        float v = xr[i];
        s += v; ss += v * v;
    }
    #pragma unroll
    for (int m = 16; m >= 1; m >>= 1) {
        s  += __shfl_xor(s,  m);
        ss += __shfl_xor(ss, m);
    }
    __shared__ float red0[8], red1[8];
    if ((threadIdx.x & 31) == 0) {
        red0[threadIdx.x >> 5] = s;
        red1[threadIdx.x >> 5] = ss;
    }
    __syncthreads();
    s = 0.f; ss = 0.f;
    #pragma unroll
    for (int w = 0; w < 8; ++w) { s += red0[w]; ss += red1[w]; }
    const float mu  = s / C;
    const float var = ss / C - mu * mu;
    const float inv = rsqrtf(var + 1e-5f);
    __bf16* orow = out + (size_t)row * C;
    for (int i = threadIdx.x; i < C; i += 256)
        orow[i] = (__bf16)((xr[i] - mu) * inv * g[i] + b[i]);
}

// ---------------------------------------------------------------------------
// B-tile staging: 32(k) x 128(n) slab of row-major B, transposed into Bt[n][k]
// ---------------------------------------------------------------------------
__device__ __forceinline__ void stage_b_tile(__bf16 (*bt)[32],
                                             const __bf16* __restrict__ B,
                                             int N, int k0, int bn, int t) {
    const int n2 = (t & 63) * 2;
    const int kk = t >> 6;                               // 0..3
    #pragma unroll
    for (int i = 0; i < 8; ++i) {
        const int k = kk * 8 + i;
        const __bf16* gb = B + (size_t)(k0 + k) * N + bn + n2;
        bt[n2][k]     = gb[0];
        bt[n2 + 1][k] = gb[1];
    }
}

// ---------------------------------------------------------------------------
// Tiled WMMA GEMM with TDM-pipelined A tiles and double-buffered LDS.
// C(MxN) = A(MxK,bf16 rm) * B(KxN,bf16 rm)
// 256 threads = 8 waves; tile 128x128; wave grid 4x2; each wave: 2x4 16x16 tiles
// EPI==0: out f32 = acc + res      (residual add)
// EPI==1: out bf16 = acc * scale
// ---------------------------------------------------------------------------
template <int EPI>
__global__ __launch_bounds__(256)
void gemm_bf16_kernel(const __bf16* __restrict__ A, const __bf16* __restrict__ B,
                      float* __restrict__ outF, __bf16* __restrict__ outH,
                      const float* __restrict__ res,
                      int M, int N, int K, float scale) {
    __shared__ __align__(16) __bf16 As[2][128][32];   // [buf][m][k]
    __shared__ __align__(16) __bf16 Bt[2][128][32];   // [buf][n][k] (transposed)

    const int t    = threadIdx.x;
    const int lane = t & 31, wave = t >> 5;
    const int wm = wave >> 1, wn = wave & 1;          // 4 x 2 wave grid
    const int half = lane >> 4, l16 = lane & 15;
    const int bm = blockIdx.y * 128, bn = blockIdx.x * 128;
    const int nk = K >> 5;

    v8f acc[2][4] = {};

    const __bf16* Abase = A + (size_t)bm * K;

    // ---- prologue: async A tile 0 via TDM, stage B tile 0
    if (wave == 0)
        tdm_load_a_tile(Abase, (unsigned)(size_t)&As[0][0][0], 128, K);
    stage_b_tile(Bt[0], B, N, 0, bn, t);

    for (int kt = 0; kt < nk; ++kt) {
        const int cur = kt & 1;
        if (wave == 0) __builtin_amdgcn_s_wait_tensorcnt(0);
        __syncthreads();                               // As[cur], Bt[cur] ready

        if (kt + 1 < nk) {                             // prefetch next tiles
            if (wave == 0)
                tdm_load_a_tile(Abase + (kt + 1) * 32,
                                (unsigned)(size_t)&As[cur ^ 1][0][0], 128, K);
            stage_b_tile(Bt[cur ^ 1], B, N, (kt + 1) * 32, bn, t);
        }

        // ---- fragments + WMMA on current buffer
        v16bf af[2], bf[4];
        #pragma unroll
        for (int ms = 0; ms < 2; ++ms) {
            const int r = wm * 32 + ms * 16 + l16;
            af[ms] = ld_frag(&As[cur][r][half * 8], &As[cur][r][half * 8 + 16]);
        }
        #pragma unroll
        for (int ns = 0; ns < 4; ++ns) {
            const int n = wn * 64 + ns * 16 + l16;
            bf[ns] = ld_frag(&Bt[cur][n][half * 8], &Bt[cur][n][half * 8 + 16]);
        }
        #pragma unroll
        for (int ms = 0; ms < 2; ++ms)
            #pragma unroll
            for (int ns = 0; ns < 4; ++ns)
                acc[ms][ns] = wmma_bf16(af[ms], bf[ns], acc[ms][ns]);
    }

    // ---- epilogue: C layout = vgpr vi -> M = vi + 8*half ; N = l16
    #pragma unroll
    for (int ms = 0; ms < 2; ++ms)
        #pragma unroll
        for (int ns = 0; ns < 4; ++ns)
            #pragma unroll
            for (int vi = 0; vi < 8; ++vi) {
                const int row = bm + wm * 32 + ms * 16 + vi + 8 * half;
                const int col = bn + wn * 64 + ns * 16 + l16;
                const float v = acc[ms][ns][vi];
                if (EPI == 0)
                    outF[(size_t)row * N + col] = v + res[(size_t)row * N + col];
                else
                    outH[(size_t)row * N + col] = (__bf16)(v * scale);
            }
}

// ---------------------------------------------------------------------------
// Fused SwiGLU dual GEMM: out = bf16( silu(A*Bg) * (A*Bu) )
// 256 threads = 8 waves; tile 128x64; wave w owns 16 rows, full 64 cols
// ---------------------------------------------------------------------------
__global__ __launch_bounds__(256)
void ffn_dual_gemm_kernel(const __bf16* __restrict__ A,
                          const __bf16* __restrict__ Bg,
                          const __bf16* __restrict__ Bu,
                          __bf16* __restrict__ out, int M, int N, int K) {
    __shared__ __align__(16) __bf16 As[128][32];
    __shared__ __align__(16) __bf16 Bgt[64][32];
    __shared__ __align__(16) __bf16 But[64][32];

    const int t = threadIdx.x;
    const int lane = t & 31, w = t >> 5;               // 8 waves, wave -> 16 rows
    const int half = lane >> 4, l16 = lane & 15;
    const int bm = blockIdx.y * 128, bn = blockIdx.x * 64;

    v8f accg[4] = {}, accu[4] = {};

    for (int k0 = 0; k0 < K; k0 += 32) {
        {   // A tile
            const int r = t >> 1;
            const __bf16* ga = A + (size_t)(bm + r) * K + k0 + (t & 1) * 16;
            f4 x0 = *(const f4*)ga;
            f4 x1 = *(const f4*)(ga + 8);
            *(f4*)&As[r][(t & 1) * 16]     = x0;
            *(f4*)&As[r][(t & 1) * 16 + 8] = x1;
        }
        {   // Bg/Bu tiles transposed: 32 k x 64 n each, 8 elems/thread/tile
            const int n2 = (t & 31) * 2;
            const int kk = t >> 5;                     // 0..7
            #pragma unroll
            for (int i = 0; i < 4; ++i) {
                const int k = kk * 4 + i;
                const __bf16* gg = Bg + (size_t)(k0 + k) * N + bn + n2;
                const __bf16* gu = Bu + (size_t)(k0 + k) * N + bn + n2;
                Bgt[n2][k] = gg[0]; Bgt[n2 + 1][k] = gg[1];
                But[n2][k] = gu[0]; But[n2 + 1][k] = gu[1];
            }
        }
        __syncthreads();

        v16bf af;
        {
            const int rr = w * 16 + l16;
            af = ld_frag(&As[rr][half * 8], &As[rr][half * 8 + 16]);
        }
        #pragma unroll
        for (int ns = 0; ns < 4; ++ns) {
            const int n = ns * 16 + l16;
            v16bf bg = ld_frag(&Bgt[n][half * 8], &Bgt[n][half * 8 + 16]);
            v16bf bu = ld_frag(&But[n][half * 8], &But[n][half * 8 + 16]);
            accg[ns] = wmma_bf16(af, bg, accg[ns]);
            accu[ns] = wmma_bf16(af, bu, accu[ns]);
        }
        __syncthreads();
    }

    #pragma unroll
    for (int ns = 0; ns < 4; ++ns)
        #pragma unroll
        for (int vi = 0; vi < 8; ++vi) {
            const int row = bm + w * 16 + vi + 8 * half;
            const int col = bn + ns * 16 + l16;
            const float g = accg[ns][vi];
            const float u = accu[ns][vi];
            const float sg = g / (1.f + __expf(-g));   // silu
            out[(size_t)row * N + col] = (__bf16)(sg * u);
        }
}

// ---------------------------------------------------------------------------
// Flash attention, causal. One block per (q-tile of 64, head, batch).
// 128 threads = 4 waves; wave w owns 16 query rows. Dk = 64.
// Q pre-scaled by 1/sqrt(Dk) at projection time.
// ---------------------------------------------------------------------------
__global__ __launch_bounds__(128)
void flash_attn_kernel(const __bf16* __restrict__ Q, const __bf16* __restrict__ K,
                       const __bf16* __restrict__ V, __bf16* __restrict__ O,
                       int T, int C) {
    __shared__ __align__(16) __bf16 Qs[64][64];   // [q][d]
    __shared__ __align__(16) __bf16 Ks[64][64];   // [key][d]  (rows = B^T cols)
    __shared__ __align__(16) __bf16 Vt[64][64];   // [d][key]  (transposed)
    __shared__ __align__(16) __bf16 Ps[64][64];   // [q][key]  bf16 probs

    const int t = threadIdx.x;
    const int lane = t & 31, w = t >> 5;          // 4 waves
    const int half = lane >> 4, l16 = lane & 15;
    const int h = blockIdx.y, b = blockIdx.z;
    const int qbase = blockIdx.x * 64;
    const size_t base = ((size_t)b * T) * C + (size_t)h * 64;

    {   // stage Q tile once
        const int r = t >> 1, dof = (t & 1) * 32;
        const __bf16* g = Q + base + (size_t)(qbase + r) * C + dof;
        #pragma unroll
        for (int i = 0; i < 4; ++i)
            *(f4*)&Qs[r][dof + i * 8] = *(const f4*)(g + i * 8);
    }

    float m[8], l[8];
    v8f o[4] = {};
    #pragma unroll
    for (int vi = 0; vi < 8; ++vi) { m[vi] = -INFINITY; l[vi] = 0.f; }

    const int qt = qbase >> 6;
    for (int kt = 0; kt <= qt; ++kt) {
        __syncthreads();
        {   // K tile (row-major: key rows double as B^T columns)
            const int r = t >> 1, dof = (t & 1) * 32;
            const __bf16* g = K + base + (size_t)(kt * 64 + r) * C + dof;
            #pragma unroll
            for (int i = 0; i < 4; ++i)
                *(f4*)&Ks[r][dof + i * 8] = *(const f4*)(g + i * 8);
        }
        {   // V tile transposed into Vt[d][key]
            const int d = (t & 31) * 2;
            #pragma unroll
            for (int p = 0; p < 16; ++p) {
                const int key = p * 4 + w;
                const __bf16* g = V + base + (size_t)(kt * 64 + key) * C + d;
                Vt[d][key]     = g[0];
                Vt[d + 1][key] = g[1];
            }
        }
        __syncthreads();

        // ---- S = Q K^T : 16 rows x 64 keys per wave (K-dim 64 = 2 WMMAs)
        v8f s[4] = {};
        #pragma unroll
        for (int c = 0; c < 2; ++c) {
            const int r = w * 16 + l16;
            v16bf qa = ld_frag(&Qs[r][c * 32 + half * 8],
                               &Qs[r][c * 32 + half * 8 + 16]);
            #pragma unroll
            for (int ns = 0; ns < 4; ++ns) {
                const int n = ns * 16 + l16;
                v16bf kb = ld_frag(&Ks[n][c * 32 + half * 8],
                                   &Ks[n][c * 32 + half * 8 + 16]);
                s[ns] = wmma_bf16(qa, kb, s[ns]);
            }
        }

        // ---- causal mask on diagonal tile
        if (kt == qt) {
            #pragma unroll
            for (int ns = 0; ns < 4; ++ns) {
                const int col = kt * 64 + ns * 16 + l16;
                #pragma unroll
                for (int vi = 0; vi < 8; ++vi) {
                    const int row = qbase + w * 16 + vi + 8 * half;
                    if (col > row) s[ns][vi] = -INFINITY;
                }
            }
        }

        // ---- online softmax (row stats reduce across 16 lanes of each half)
        #pragma unroll
        for (int vi = 0; vi < 8; ++vi) {
            float rmax = s[0][vi];
            #pragma unroll
            for (int ns = 1; ns < 4; ++ns) rmax = fmaxf(rmax, s[ns][vi]);
            #pragma unroll
            for (int msk = 8; msk >= 1; msk >>= 1)
                rmax = fmaxf(rmax, __shfl_xor(rmax, msk));
            const float mnew = fmaxf(m[vi], rmax);
            const float corr = __expf(m[vi] - mnew);
            float rsum = 0.f;
            #pragma unroll
            for (int ns = 0; ns < 4; ++ns) {
                const float p = __expf(s[ns][vi] - mnew);
                s[ns][vi] = p;
                rsum += p;
            }
            #pragma unroll
            for (int msk = 8; msk >= 1; msk >>= 1)
                rsum += __shfl_xor(rsum, msk);
            l[vi] = l[vi] * corr + rsum;
            m[vi] = mnew;
            #pragma unroll
            for (int ns = 0; ns < 4; ++ns) o[ns][vi] *= corr;
        }

        // ---- stage P (bf16) into this wave's private rows of Ps
        #pragma unroll
        for (int ns = 0; ns < 4; ++ns)
            #pragma unroll
            for (int vi = 0; vi < 8; ++vi)
                Ps[w * 16 + vi + 8 * half][ns * 16 + l16] = (__bf16)s[ns][vi];

        // ---- O += P V  (keys = contraction dim, 2 WMMAs of K=32)
        #pragma unroll
        for (int c = 0; c < 2; ++c) {
            const int r = w * 16 + l16;
            v16bf pa = ld_frag(&Ps[r][c * 32 + half * 8],
                               &Ps[r][c * 32 + half * 8 + 16]);
            #pragma unroll
            for (int ns = 0; ns < 4; ++ns) {
                const int d = ns * 16 + l16;
                v16bf vb = ld_frag(&Vt[d][c * 32 + half * 8],
                                   &Vt[d][c * 32 + half * 8 + 16]);
                o[ns] = wmma_bf16(pa, vb, o[ns]);
            }
        }
    }

    // ---- normalize + store bf16, interleaved-head layout [b][t][h*64+d]
    #pragma unroll
    for (int ns = 0; ns < 4; ++ns)
        #pragma unroll
        for (int vi = 0; vi < 8; ++vi) {
            const int row = qbase + w * 16 + vi + 8 * half;
            const int d   = ns * 16 + l16;
            O[base + (size_t)row * C + d] = (__bf16)(o[ns][vi] / l[vi]);
        }
}

// ---------------------------------------------------------------------------
// Host-side orchestration
// ---------------------------------------------------------------------------
extern "C" void kernel_launch(void* const* d_in, const int* in_sizes, int n_in,
                              void* d_out, int out_size, void* d_ws, size_t ws_size,
                              hipStream_t stream) {
    const float* x      = (const float*)d_in[0];
    const float* attn_g = (const float*)d_in[1];
    const float* attn_b = (const float*)d_in[2];
    const float* Wq     = (const float*)d_in[3];
    const float* Wk     = (const float*)d_in[4];
    const float* Wv     = (const float*)d_in[5];
    const float* Wo     = (const float*)d_in[6];
    const float* ffn_g  = (const float*)d_in[7];
    const float* ffn_b  = (const float*)d_in[8];
    const float* W1     = (const float*)d_in[9];
    const float* Wg     = (const float*)d_in[10];
    const float* W2     = (const float*)d_in[11];

    const int Bz = 2, T = 2048, C = 1024, M = Bz * T, F = 2816;

    char* ws = (char*)d_ws;
    size_t off = 0;
    auto alloc = [&](size_t bytes) -> char* {
        char* p = ws + off;
        off = (off + bytes + 255) & ~(size_t)255;
        return p;
    };

    __bf16* Wqb  = (__bf16*)alloc((size_t)C * C * 2);
    __bf16* Wkb  = (__bf16*)alloc((size_t)C * C * 2);
    __bf16* Wvb  = (__bf16*)alloc((size_t)C * C * 2);
    __bf16* Wob  = (__bf16*)alloc((size_t)C * C * 2);
    __bf16* Wgb  = (__bf16*)alloc((size_t)C * F * 2);
    __bf16* W1b  = (__bf16*)alloc((size_t)C * F * 2);
    __bf16* W2b  = (__bf16*)alloc((size_t)F * C * 2);
    __bf16* n1b  = (__bf16*)alloc((size_t)M * C * 2);
    __bf16* Qb   = (__bf16*)alloc((size_t)M * C * 2);
    __bf16* Kb   = (__bf16*)alloc((size_t)M * C * 2);
    __bf16* Vb   = (__bf16*)alloc((size_t)M * C * 2);
    __bf16* Ab   = (__bf16*)alloc((size_t)M * C * 2);   // attention out (bf16)
    float*  x1f  = (float*) alloc((size_t)M * C * 4);   // x + attn@Wo (f32)
    __bf16* n2b  = (__bf16*)alloc((size_t)M * C * 2);
    __bf16* hb   = (__bf16*)alloc((size_t)M * F * 2);   // silu(g)*u (bf16)

    // 1. weights -> bf16
    cvt_f32_bf16_kernel<<<512, 256, 0, stream>>>(Wq, Wqb, C * C);
    cvt_f32_bf16_kernel<<<512, 256, 0, stream>>>(Wk, Wkb, C * C);
    cvt_f32_bf16_kernel<<<512, 256, 0, stream>>>(Wv, Wvb, C * C);
    cvt_f32_bf16_kernel<<<512, 256, 0, stream>>>(Wo, Wob, C * C);
    cvt_f32_bf16_kernel<<<512, 256, 0, stream>>>(Wg, Wgb, C * F);
    cvt_f32_bf16_kernel<<<512, 256, 0, stream>>>(W1, W1b, C * F);
    cvt_f32_bf16_kernel<<<512, 256, 0, stream>>>(W2, W2b, F * C);

    // 2. LN1
    layernorm_bf16_kernel<<<M, 256, 0, stream>>>(x, attn_g, attn_b, n1b, C);

    // 3. Q/K/V projections (Q pre-scaled by 1/sqrt(64) = 0.125)
    dim3 gqkv(C / 128, M / 128);
    gemm_bf16_kernel<1><<<gqkv, 256, 0, stream>>>(n1b, Wqb, nullptr, Qb, nullptr,
                                                  M, C, C, 0.125f);
    gemm_bf16_kernel<1><<<gqkv, 256, 0, stream>>>(n1b, Wkb, nullptr, Kb, nullptr,
                                                  M, C, C, 1.0f);
    gemm_bf16_kernel<1><<<gqkv, 256, 0, stream>>>(n1b, Wvb, nullptr, Vb, nullptr,
                                                  M, C, C, 1.0f);

    // 4. causal flash attention
    flash_attn_kernel<<<dim3(T / 64, 16, Bz), 128, 0, stream>>>(Qb, Kb, Vb, Ab, T, C);

    // 5. Wo projection + residual -> x1 (f32)
    gemm_bf16_kernel<0><<<gqkv, 256, 0, stream>>>(Ab, Wob, x1f, nullptr, x,
                                                  M, C, C, 1.0f);

    // 6. LN2
    layernorm_bf16_kernel<<<M, 256, 0, stream>>>(x1f, ffn_g, ffn_b, n2b, C);

    // 7. fused SwiGLU dual GEMM -> hb (bf16)
    ffn_dual_gemm_kernel<<<dim3(F / 64, M / 128), 256, 0, stream>>>(n2b, Wgb, W1b,
                                                                    hb, M, F, C);

    // 8. hb @ W2 + residual -> d_out (f32)
    gemm_bf16_kernel<0><<<dim3(C / 128, M / 128), 256, 0, stream>>>(
        hb, W2b, (float*)d_out, nullptr, x1f, M, C, F, 1.0f);

    (void)in_sizes; (void)n_in; (void)out_size; (void)ws_size;
}
